// GATModel_63221918597529
// MI455X (gfx1250) — matrix operand order repaired and checked by reading, not compiled, not run on hardware
//
#include <hip/hip_runtime.h>
#include <hip/hip_bf16.h>
#include <math.h>

// ---------------------------------------------------------------------------
// GATv2 (4 layers, HIDDEN=128, HEADS=4) forward on gfx1250.
// GEMMs via v_wmma_f32_16x16x32_bf16; A-operands pre-produced in bf16 row-major
// (zero in-loop conversion); B-operands pre-swizzled and staged into LDS by the
// Tensor Data Mover (tensor_load_to_lds + s_wait_tensorcnt), cooperative-copy
// fallback if the builtin is unavailable.
// All intermediates live in d_ws (~165MB) -> resident in MI455X 192MB L2.
// ---------------------------------------------------------------------------

#define N_NODES 50000
#define N_EDGES 800000
#define EA_TOT  (N_EDGES + N_NODES)   // with self loops
#define HIDDEN  128
#define STRIPS  (N_NODES / 16)        // 3125, exact
#define N_GRAPHS 64
#define N_CLASSES 10

typedef __attribute__((ext_vector_type(16))) __bf16 v16bf;
typedef __attribute__((ext_vector_type(8)))  float  v8f;
typedef unsigned short ushort_t;

union Frag16 { uint4 q[2]; v16bf v; };   // 16 bf16 = 32B

__device__ __forceinline__ unsigned pkbf(float a, float b) {
    union { __bf16 h[2]; unsigned u; } t;
    t.h[0] = (__bf16)a; t.h[1] = (__bf16)b;   // hardware v_cvt path
    return t.u;
}
__device__ __forceinline__ ushort_t f2bf(float f) {
    union { __bf16 h; ushort_t u; } t;
    t.h = (__bf16)f;
    return t.u;
}
__device__ __forceinline__ float gelu_f(float x) {
    return 0.5f * x * (1.0f + erff(x * 0.70710678118654752f));
}
__device__ __forceinline__ float lrelu(float x) { return x > 0.0f ? x : 0.2f * x; }

// orderable-uint encoding for float atomic max
__device__ __forceinline__ unsigned fenc(float f) {
    unsigned u = __float_as_uint(f);
    return (u & 0x80000000u) ? ~u : (u | 0x80000000u);
}
__device__ __forceinline__ float fdec(unsigned e) {
    unsigned u = (e & 0x80000000u) ? (e & 0x7FFFFFFFu) : ~e;
    return __uint_as_float(u);
}

// ---------------------------------------------------------------------------
// TDM: DMA a 2D tile of B-fragments (nrows x 8192B, row stride rstride_units x 8B)
// from global into LDS at byte offset lds_off.  D# packing per ISA 08 §8.
// ---------------------------------------------------------------------------
#if __has_builtin(__builtin_amdgcn_tensor_load_to_lds)
#define HAVE_TDM 1
__device__ __forceinline__ void tdm_load_b(const ushort_t* src, unsigned lds_off,
                                           int nrows, unsigned rstride_units) {
    typedef __attribute__((ext_vector_type(4))) unsigned u32x4;
    typedef __attribute__((ext_vector_type(8))) int i32x8;
    typedef __attribute__((ext_vector_type(4))) int i32x4;
    unsigned long long ga = (unsigned long long)src;
    u32x4 g0;
    g0[0] = 1u;                                              // count=1 (valid user D#)
    g0[1] = lds_off;                                         // lds_addr (bytes)
    g0[2] = (unsigned)ga;                                    // global_addr[31:0]
    g0[3] = (unsigned)((ga >> 32) & 0x01FFFFFFull) | (2u << 30); // addr[56:32] | type=2
    // data_size=3 (8B units); row = 8192B = 1024 units
    i32x8 g1;
    g1[0] = (int)(3u << 16);                                 // data_size
    g1[1] = (int)(1024u << 16);                              // tensor_dim0[15:0] @ bits 63:48
    g1[2] = (int)(((unsigned)nrows) << 16);                  // dim0 hi=0 | tensor_dim1[15:0]
    g1[3] = (int)(1024u << 16);                              // dim1 hi=0 | tile_dim0=1024
    g1[4] = (int)(nrows & 0xFFFF);                           // tile_dim1 (tile_dim2=0)
    g1[5] = (int)rstride_units;                              // tensor_dim0_stride[31:0]
    g1[6] = 0;
    g1[7] = 0;
    i32x4 gz = {0, 0, 0, 0};
#if defined(__clang_major__) && __clang_major__ >= 23
    i32x8 gz8 = {0, 0, 0, 0, 0, 0, 0, 0};
    __builtin_amdgcn_tensor_load_to_lds(g0, g1, gz, gz, gz8, 0);
#else
    __builtin_amdgcn_tensor_load_to_lds(g0, g1, gz, gz, 0);
#endif
}
#else
#define HAVE_TDM 0
#endif

// ---------------------------------------------------------------------------
// Weight pre-swizzle: f32 [K x NC] row-major -> bf16 B-fragments.
// Fragment f = kstep*(NC/16)+nt, 32 lanes x 16 halves:
//   lane l(<16): col nt*16+l,      K = kstep*32 + e
//   lane l(>=16): col nt*16+l-16,  K = kstep*32 + 16 + e
// ---------------------------------------------------------------------------
__global__ void k_swizzle(const float* __restrict__ W, ushort_t* __restrict__ out,
                          int K, int NC) {
    int idx = blockIdx.x * blockDim.x + threadIdx.x;
    if (idx >= K * NC) return;
    int f = idx >> 9;
    int r = idx & 511;
    int l = r >> 4;
    int e = r & 15;
    int ntiles = NC >> 4;
    int kstep = f / ntiles;
    int nt    = f - kstep * ntiles;
    int col = nt * 16 + (l & 15);
    int k   = kstep * 32 + (l & 16) + e;
    out[idx] = f2bf(W[k * NC + col]);
}

// row-major f32 -> bf16 (for the input features x)
__global__ void k_f32_to_bf16(const float* __restrict__ in, ushort_t* __restrict__ out,
                              long long n4) {
    long long i = blockIdx.x * (long long)blockDim.x + threadIdx.x;
    if (i >= n4) return;
    float4 v = ((const float4*)in)[i];
    uint2 o; o.x = pkbf(v.x, v.y); o.y = pkbf(v.z, v.w);
    ((uint2*)out)[i] = o;
}

// ---------------------------------------------------------------------------
// WMMA GEMM: one wave computes a 16-row x 128-col strip of C = A @ B + bias.
//   A: bf16 row-major [M x K] -> A-fragments are two global_load_b128, no cvt
//   Bswz: pre-swizzled bf16 fragments; block's 8 n-tiles staged in LDS via TDM
//   EPI: 1 = bias->f32, 2 = bias+GELU->bf16, 3 = bias+residual->f32
// ---------------------------------------------------------------------------
template <int K, int EPI>
__global__ __launch_bounds__(256)
void k_gemm(const ushort_t* __restrict__ A, const ushort_t* __restrict__ Bswz,
            int frag_stride, int nt_base, const float* __restrict__ bias,
            void* __restrict__ Cout, const float* __restrict__ resid,
            int ldc, int col_off, int strips) {
    constexpr int KSTEPS = K / 32;
    constexpr int FRAGS  = KSTEPS * 8;
    __shared__ ushort_t smem[FRAGS * 512];          // <= 64KB of 320KB/WGP

#if HAVE_TDM
    if (threadIdx.x < 32) {                          // wave 0 drives the DMA
        tdm_load_b(Bswz + (size_t)nt_base * 512,
                   (unsigned)(size_t)&smem[0],
                   KSTEPS, (unsigned)frag_stride * 128u);
        __builtin_amdgcn_s_wait_tensorcnt(0);
    }
#else
    for (int c = threadIdx.x; c < FRAGS * 64; c += blockDim.x) {  // 16B chunks
        int fi = c >> 6;
        int ks = fi >> 3, nt = fi & 7;
        int within = c & 63;
        const uint4* src = (const uint4*)(Bswz +
            (size_t)(ks * frag_stride + nt_base + nt) * 512) + within;
        ((uint4*)smem)[c] = *src;
    }
#endif
    __syncthreads();

    int wid  = blockIdx.x * (blockDim.x >> 5) + (threadIdx.x >> 5);
    if (wid >= strips) return;                 // whole wave exits together
    int lane = threadIdx.x & 31;
    int row0 = wid * 16;

    v8f acc[8];
#pragma unroll
    for (int i = 0; i < 8; ++i) acc[i] = v8f{0.f,0.f,0.f,0.f,0.f,0.f,0.f,0.f};

    const int arow = row0 + (lane & 15);
    const int kofs = (lane & 16) ? 16 : 0;

#pragma unroll
    for (int ks = 0; ks < KSTEPS; ++ks) {
        // A-fragment: elements 0..15 = K kbase+kofs.. contiguous (matches Bswz)
        const ushort_t* ap = A + (size_t)arow * K + ks * 32 + kofs;
        Frag16 a;
        a.q[0] = *(const uint4*)(ap);
        a.q[1] = *(const uint4*)(ap + 8);
        // batch all 8 B-fragments into distinct registers, then 8 WMMAs
        Frag16 b[8];
#pragma unroll
        for (int nt = 0; nt < 8; ++nt) {
            const uint4* bp = (const uint4*)(smem + (ks * 8 + nt) * 512) + lane * 2;
            b[nt].q[0] = bp[0];
            b[nt].q[1] = bp[1];
        }
#pragma unroll
        for (int nt = 0; nt < 8; ++nt) {
            acc[nt] = __builtin_amdgcn_wmma_f32_16x16x32_bf16(
                false, a.v, false, b[nt].v, (short)0, acc[nt], false, false);
        }
    }

    // D layout: element r of lane n(<16) = D[r][n]; lane n(>=16) = D[8+r][n-16]
    int n    = lane & 15;
    int rofs = (lane & 16) ? 8 : 0;
#pragma unroll
    for (int nt = 0; nt < 8; ++nt) {
        int colg = col_off + nt * 16 + n;
        float bv = bias[colg];
#pragma unroll
        for (int r = 0; r < 8; ++r) {
            int row = row0 + rofs + r;
            float v = acc[nt][r] + bv;
            if (EPI == 2) {
                ((ushort_t*)Cout)[(size_t)row * ldc + colg] = f2bf(gelu_f(v));
            } else {
                if (EPI == 3) v += resid[(size_t)row * ldc + colg];
                ((float*)Cout)[(size_t)row * ldc + colg] = v;
            }
        }
    }
}

// ---------------------------------------------------------------------------
// Row-wise LayerNorm over 128 channels (wave per row).
// GELU optional; output f32 (BF16OUT=0) or bf16 row-major (BF16OUT=1).
// ---------------------------------------------------------------------------
template <int GELU, int BF16OUT>
__global__ __launch_bounds__(256)
void k_layernorm(const float* __restrict__ in, void* __restrict__ out,
                 const float* __restrict__ g, const float* __restrict__ b, int rows) {
    int row  = blockIdx.x * (blockDim.x >> 5) + (threadIdx.x >> 5);
    if (row >= rows) return;
    int lane = threadIdx.x & 31;
    float4 v = *(const float4*)(in + (size_t)row * HIDDEN + lane * 4);
    float s = v.x + v.y + v.z + v.w;
#pragma unroll
    for (int o = 16; o > 0; o >>= 1) s += __shfl_xor(s, o, 32);
    float mu = s * (1.0f / 128.0f);
    float4 d = {v.x - mu, v.y - mu, v.z - mu, v.w - mu};
    float q = d.x * d.x + d.y * d.y + d.z * d.z + d.w * d.w;
#pragma unroll
    for (int o = 16; o > 0; o >>= 1) q += __shfl_xor(q, o, 32);
    float rstd = rsqrtf(q * (1.0f / 128.0f) + 1e-5f);
    float4 gv = *(const float4*)(g + lane * 4);
    float4 bv = *(const float4*)(b + lane * 4);
    float4 o4;
    o4.x = d.x * rstd * gv.x + bv.x;
    o4.y = d.y * rstd * gv.y + bv.y;
    o4.z = d.z * rstd * gv.z + bv.z;
    o4.w = d.w * rstd * gv.w + bv.w;
    if (GELU) { o4.x = gelu_f(o4.x); o4.y = gelu_f(o4.y); o4.z = gelu_f(o4.z); o4.w = gelu_f(o4.w); }
    if (BF16OUT) {
        uint2 o; o.x = pkbf(o4.x, o4.y); o.y = pkbf(o4.z, o4.w);
        ((uint2*)((ushort_t*)out + (size_t)row * HIDDEN))[lane] = o;
    } else {
        *(float4*)((float*)out + (size_t)row * HIDDEN + lane * 4) = o4;
    }
}

// ---------------------------------------------------------------------------
// Edge pass 1: logits[e,h] = dot(leaky_relu(xl[src]+xr[dst]), att[h]); seg-max.
// ---------------------------------------------------------------------------
__global__ __launch_bounds__(256)
void k_edge_logits(const int* __restrict__ ei, const float* __restrict__ xl,
                   const float* __restrict__ xr, const float* __restrict__ att,
                   float* __restrict__ logits, unsigned* __restrict__ mx) {
    int w = blockIdx.x * (blockDim.x >> 5) + (threadIdx.x >> 5);
    if (w >= EA_TOT) return;
    int lane = threadIdx.x & 31;
    int s, d;
    if (w < N_EDGES) { s = ei[w]; d = ei[N_EDGES + w]; }
    else             { s = w - N_EDGES; d = s; }
    float4 a = *(const float4*)(xl + (size_t)s * HIDDEN + lane * 4);
    float4 b = *(const float4*)(xr + (size_t)d * HIDDEN + lane * 4);
    float4 t = *(const float4*)(att + lane * 4);   // att[i] is [4,32] = flat 128
    float p = lrelu(a.x + b.x) * t.x + lrelu(a.y + b.y) * t.y +
              lrelu(a.z + b.z) * t.z + lrelu(a.w + b.w) * t.w;
    p += __shfl_xor(p, 1, 32);
    p += __shfl_xor(p, 2, 32);
    p += __shfl_xor(p, 4, 32);                     // reduce within 8-lane head group
    if ((lane & 7) == 0) {
        int h = lane >> 3;
        logits[(size_t)w * 4 + h] = p;
        atomicMax(mx + (size_t)d * 4 + h, fenc(p));
    }
}

// Edge pass 2: ex = exp(logit - max[dst]); den[dst] += ex.
__global__ void k_edge_exp(const int* __restrict__ ei, float* __restrict__ logits,
                           const unsigned* __restrict__ mx, float* __restrict__ den) {
    int t = blockIdx.x * blockDim.x + threadIdx.x;
    if (t >= EA_TOT * 4) return;
    int e = t >> 2, h = t & 3;
    int d = (e < N_EDGES) ? ei[N_EDGES + e] : e - N_EDGES;
    float x = expf(logits[t] - fdec(mx[(size_t)d * 4 + h]));
    logits[t] = x;
    atomicAdd(den + (size_t)d * 4 + h, x);
}

// Edge pass 3: out[dst] += xl[src] * (ex/den[dst]).
__global__ __launch_bounds__(256)
void k_edge_aggr(const int* __restrict__ ei, const float* __restrict__ xl,
                 const float* __restrict__ ex, const float* __restrict__ den,
                 float* __restrict__ outacc) {
    int w = blockIdx.x * (blockDim.x >> 5) + (threadIdx.x >> 5);
    if (w >= EA_TOT) return;
    int lane = threadIdx.x & 31;
    int s, d;
    if (w < N_EDGES) { s = ei[w]; d = ei[N_EDGES + w]; }
    else             { s = w - N_EDGES; d = s; }
    int h = lane >> 3;
    float alpha = ex[(size_t)w * 4 + h] / den[(size_t)d * 4 + h];
    float4 a = *(const float4*)(xl + (size_t)s * HIDDEN + lane * 4);
    float* o = outacc + (size_t)d * HIDDEN + lane * 4;
    atomicAdd(o + 0, a.x * alpha);
    atomicAdd(o + 1, a.y * alpha);
    atomicAdd(o + 2, a.z * alpha);
    atomicAdd(o + 3, a.w * alpha);
}

// h += attout + conv_b (float4 per thread)
__global__ void k_add_attn(float* __restrict__ h, const float* __restrict__ ao,
                           const float* __restrict__ cb, int n4) {
    int i = blockIdx.x * blockDim.x + threadIdx.x;
    if (i >= n4) return;
    int c4 = i & 31;
    float4 hv = ((const float4*)h)[i];
    float4 av = ((const float4*)ao)[i];
    float4 bv = ((const float4*)cb)[c4];
    hv.x += av.x + bv.x; hv.y += av.y + bv.y;
    hv.z += av.z + bv.z; hv.w += av.w + bv.w;
    ((float4*)h)[i] = hv;
}

__global__ void k_fill(unsigned* __restrict__ p, unsigned v, long long n) {
    long long i = blockIdx.x * (long long)blockDim.x + threadIdx.x;
    if (i < n) p[i] = v;
}

// per-graph mean pool accumulation
__global__ void k_pool(const float* __restrict__ h, const int* __restrict__ batch,
                       float* __restrict__ sums, float* __restrict__ cnt) {
    int t = blockIdx.x * blockDim.x + threadIdx.x;
    if (t >= N_NODES * 32) return;
    int node = t >> 5, c4 = t & 31;
    int g = batch[node];
    float4 v = *(const float4*)(h + (size_t)node * HIDDEN + c4 * 4);
    float* s = sums + (size_t)g * HIDDEN + c4 * 4;
    atomicAdd(s + 0, v.x); atomicAdd(s + 1, v.y);
    atomicAdd(s + 2, v.z); atomicAdd(s + 3, v.w);
    if (c4 == 0) atomicAdd(cnt + g, 1.0f);
}

__global__ void k_classify(const float* __restrict__ sums, const float* __restrict__ cnt,
                           const float* __restrict__ W, const float* __restrict__ b,
                           float* __restrict__ out) {
    int t = blockIdx.x * blockDim.x + threadIdx.x;
    if (t >= N_GRAPHS * N_CLASSES) return;
    int g = t / N_CLASSES, cls = t - g * N_CLASSES;
    float c = fmaxf(cnt[g], 1.0f);
    float acc = b[cls];
    for (int k = 0; k < HIDDEN; ++k)
        acc += (sums[(size_t)g * HIDDEN + k] / c) * W[k * N_CLASSES + cls];
    out[t] = acc;
}

// ---------------------------------------------------------------------------
#define CDIV(a, b) (((a) + (b) - 1) / (b))

extern "C" void kernel_launch(void* const* d_in, const int* in_sizes, int n_in,
                              void* d_out, int out_size, void* d_ws, size_t ws_size,
                              hipStream_t stream) {
    const float* x       = (const float*)d_in[0];
    const int*   ei      = (const int*)d_in[1];
    const int*   batch   = (const int*)d_in[2];
    const float* in_W    = (const float*)d_in[3];
    const float* in_b    = (const float*)d_in[4];
    const float* in_ln_g = (const float*)d_in[5];
    const float* in_ln_b = (const float*)d_in[6];
    const float* ln_g    = (const float*)d_in[7];
    const float* ln_b    = (const float*)d_in[8];
    const float* Wl      = (const float*)d_in[9];
    const float* bl      = (const float*)d_in[10];
    const float* Wr      = (const float*)d_in[11];
    const float* br      = (const float*)d_in[12];
    const float* att     = (const float*)d_in[13];
    const float* conv_b  = (const float*)d_in[14];
    const float* ff_ln_g = (const float*)d_in[15];
    const float* ff_ln_b = (const float*)d_in[16];
    const float* ffW1    = (const float*)d_in[17];
    const float* ffb1    = (const float*)d_in[18];
    const float* ffW2    = (const float*)d_in[19];
    const float* ffb2    = (const float*)d_in[20];
    const float* clfW    = (const float*)d_in[21];
    const float* clfb    = (const float*)d_in[22];

    // ---- workspace carve-up (256B aligned; ~165 MB total, L2 resident) ----
    char* w = (char*)d_ws;
    size_t off = 0;
    auto take = [&](size_t bytes) { size_t r = off; off = (off + bytes + 255) & ~(size_t)255; return r; };
    const size_t NF = (size_t)N_NODES * HIDDEN;
    float*    h      = (float*)(w + take(NF * 4));
    float*    xl     = (float*)(w + take(NF * 4));           // also input-proj f32 tmp
    float*    xr     = (float*)(w + take(NF * 4));
    float*    attout = (float*)(w + take(NF * 4));
    ushort_t* hnb    = (ushort_t*)(w + take(NF * 2));        // LN output, bf16
    ushort_t* gb     = (ushort_t*)(w + take((size_t)N_NODES * 256 * 2)); // FFN hidden, bf16
    ushort_t* xb     = (ushort_t*)(w + take((size_t)N_NODES * 64 * 2));  // x in bf16
    float*    logits = (float*)(w + take((size_t)EA_TOT * 4 * 4));
    unsigned* mx     = (unsigned*)(w + take((size_t)N_NODES * 4 * 4));
    float*    den    = (float*)(w + take((size_t)N_NODES * 4 * 4));
    float*    sums   = (float*)(w + take((size_t)N_GRAPHS * HIDDEN * 4));
    float*    cnt    = (float*)(w + take((size_t)N_GRAPHS * 4));
    ushort_t* s_in   = (ushort_t*)(w + take((size_t)64 * 128 * 2));
    ushort_t* s_wl   = (ushort_t*)(w + take((size_t)4 * 128 * 128 * 2));
    ushort_t* s_wr   = (ushort_t*)(w + take((size_t)4 * 128 * 128 * 2));
    ushort_t* s_w1   = (ushort_t*)(w + take((size_t)4 * 128 * 256 * 2));
    ushort_t* s_w2   = (ushort_t*)(w + take((size_t)4 * 256 * 128 * 2));

    const dim3 B256(256);

    // ---- pre-swizzle weights; convert x to bf16 ----
    k_swizzle<<<CDIV(64 * 128, 256), B256, 0, stream>>>(in_W, s_in, 64, 128);
    for (int i = 0; i < 4; ++i) {
        k_swizzle<<<CDIV(128 * 128, 256), B256, 0, stream>>>(Wl + i * 16384, s_wl + i * 16384, 128, 128);
        k_swizzle<<<CDIV(128 * 128, 256), B256, 0, stream>>>(Wr + i * 16384, s_wr + i * 16384, 128, 128);
        k_swizzle<<<CDIV(128 * 256, 256), B256, 0, stream>>>(ffW1 + i * 32768, s_w1 + i * 32768, 128, 256);
        k_swizzle<<<CDIV(256 * 128, 256), B256, 0, stream>>>(ffW2 + i * 32768, s_w2 + i * 32768, 256, 128);
    }
    k_f32_to_bf16<<<CDIV(N_NODES * 16, 256), B256, 0, stream>>>(x, xb, (long long)N_NODES * 16);

    const int gemm_blocks = CDIV(STRIPS, 8);
    const int ln_blocks   = CDIV(N_NODES, 8);
    const int edge_blocks = CDIV(EA_TOT, 8);

    // ---- input projection: h = gelu(LN(x @ in_W + in_b)) ----
    k_gemm<64, 1><<<gemm_blocks, B256, 0, stream>>>(xb, s_in, 8, 0, in_b, xl, nullptr, 128, 0, STRIPS);
    k_layernorm<1, 0><<<ln_blocks, B256, 0, stream>>>(xl, h, in_ln_g, in_ln_b, N_NODES);

    for (int i = 0; i < 4; ++i) {
        // pre-attention LN -> bf16 GEMM operand
        k_layernorm<0, 1><<<ln_blocks, B256, 0, stream>>>(h, hnb, ln_g + i * 128, ln_b + i * 128, N_NODES);
        k_gemm<128, 1><<<gemm_blocks, B256, 0, stream>>>(hnb, s_wl + i * 16384, 8, 0, bl + i * 128, xl, nullptr, 128, 0, STRIPS);
        k_gemm<128, 1><<<gemm_blocks, B256, 0, stream>>>(hnb, s_wr + i * 16384, 8, 0, br + i * 128, xr, nullptr, 128, 0, STRIPS);
        // zero segment accumulators
        k_fill<<<CDIV(N_NODES * 4, 256), B256, 0, stream>>>(mx, 0u, (long long)N_NODES * 4);
        k_fill<<<CDIV(N_NODES * 4, 256), B256, 0, stream>>>((unsigned*)den, 0u, (long long)N_NODES * 4);
        k_fill<<<CDIV((long long)NF, 256), B256, 0, stream>>>((unsigned*)attout, 0u, (long long)NF);
        // attention: segment-softmax + scatter aggregate
        k_edge_logits<<<edge_blocks, B256, 0, stream>>>(ei, xl, xr, att + i * 128, logits, mx);
        k_edge_exp<<<CDIV(EA_TOT * 4, 256), B256, 0, stream>>>(ei, logits, mx, den);
        k_edge_aggr<<<edge_blocks, B256, 0, stream>>>(ei, xl, logits, den, attout);
        // h = h + attout + conv_b
        k_add_attn<<<CDIV(N_NODES * 32, 256), B256, 0, stream>>>(h, attout, conv_b + i * 128, N_NODES * 32);
        // FFN: h += W2 @ gelu(W1 @ LN(h));  FFN hidden kept in bf16
        k_layernorm<0, 1><<<ln_blocks, B256, 0, stream>>>(h, hnb, ff_ln_g + i * 128, ff_ln_b + i * 128, N_NODES);
        k_gemm<128, 2><<<gemm_blocks, B256, 0, stream>>>(hnb, s_w1 + i * 32768, 16, 0, ffb1 + i * 256, gb, nullptr, 256, 0, STRIPS);
        k_gemm<128, 2><<<gemm_blocks, B256, 0, stream>>>(hnb, s_w1 + i * 32768, 16, 8, ffb1 + i * 256, gb, nullptr, 256, 128, STRIPS);
        k_gemm<256, 3><<<gemm_blocks, B256, 0, stream>>>(gb, s_w2 + i * 32768, 8, 0, ffb2 + i * 128, h, h, 128, 0, STRIPS);
    }

    // ---- global mean pool + classifier ----
    k_fill<<<CDIV(N_GRAPHS * HIDDEN, 256), B256, 0, stream>>>((unsigned*)sums, 0u, N_GRAPHS * HIDDEN);
    k_fill<<<1, B256, 0, stream>>>((unsigned*)cnt, 0u, N_GRAPHS);
    k_pool<<<CDIV(N_NODES * 32, 256), B256, 0, stream>>>(h, batch, sums, cnt);
    k_classify<<<CDIV(N_GRAPHS * N_CLASSES, 256), B256, 0, stream>>>(sums, cnt, clfW, clfb, (float*)d_out);
}
